// CumsumAttention_45629732553257
// MI455X (gfx1250) — matrix operand order repaired and checked by reading, not compiled, not run on hardware
//
#include <hip/hip_runtime.h>
#include <hip/hip_bf16.h>

// CumsumAttention for MI455X (gfx1250, wave32, WMMA bf16).
// B=4, T=4096, C=512, VOCAB=50257.
// softplus attention has no normalization -> pure streaming accumulation.

typedef __bf16 bf16;
typedef __attribute__((ext_vector_type(16))) __bf16 bf16x16;
typedef __attribute__((ext_vector_type(8)))  __bf16 bf16x8;
typedef __attribute__((ext_vector_type(8)))  float  f32x8;

#define B_  4
#define T_  4096
#define C_  512
#define C2_ 256

static __device__ __forceinline__ f32x8 wmma_bf16(bf16x16 a, bf16x16 b, f32x8 c) {
  // D = A(16x32 bf16) * B(32x16 bf16) + C(16x16 f32)
  return __builtin_amdgcn_wmma_f32_16x16x32_bf16(false, a, false, b, (short)0, c,
                                                 false, false);
}

// ---- CDNA5 async global->LDS copy (ASYNCcnt path), 16 bytes per call -------
static __device__ __forceinline__ void async_ld_b128(void* lds, const void* g) {
#if defined(__gfx1250__)
  // vdst = 32-bit LDS byte address (generic shared ptr low bits), vaddr = 64-bit global
  asm volatile("global_load_async_to_lds_b128 %0, %1, off"
               :: "v"((unsigned int)(unsigned long long)lds), "v"(g)
               : "memory");
#else
  *(uint4*)lds = *(const uint4*)g;   // host-pass placeholder (never executed)
#endif
}
static __device__ __forceinline__ void async_wait0() {
#if defined(__gfx1250__)
  asm volatile("s_wait_asynccnt 0" ::: "memory");
#endif
}

// A fragment (16x32 bf16) from row-major [M][K] storage, leading dim ld (elems).
// Contiguous-k layout used here (not the ISA table order): lanes 0-15 hold M=lane, K 0..15; lanes 16-31: K 16..31.
static __device__ __forceinline__ bf16x16 load_a_frag(const bf16* base, int ld) {
  const int lane = threadIdx.x & 31;
  const int row  = lane & 15;
  const int kh   = (lane >> 4) << 4;      // 0 or 16
  const bf16* p  = base + row * ld + kh;
  bf16x8 lo = *(const bf16x8*)(p);
  bf16x8 hi = *(const bf16x8*)(p + 8);
  bf16x16 r;
#pragma unroll
  for (int i = 0; i < 8; ++i) { r[i] = lo[i]; r[i + 8] = hi[i]; }
  return r;
}

// B fragment (32x16 bf16, K x N) from N-major storage Bsrc[n][k], leading dim ld.
// Contiguous-k layout used here (not the ISA table order): lane&15 = N, lanes 0-15 hold K 0..15, lanes 16-31 hold K 16..31.
static __device__ __forceinline__ bf16x16 load_b_frag(const bf16* base, int ld) {
  const int lane = threadIdx.x & 31;
  const int n  = lane & 15;
  const int kb = (lane >> 4) << 4;        // 0 or 16
  return *(const bf16x16*)(base + (size_t)n * ld + kb);
}

static __device__ __forceinline__ float softplusf(float x) {
  return (x > 20.f) ? x : log1pf(expf(x));
}

// ---------------- kernel 0: weights -> bf16, transposed (n-major) -----------
__global__ void k_prep_weights(const float* __restrict__ Wk, const float* __restrict__ Wq,
                               const float* __restrict__ Wv, const float* __restrict__ Wp,
                               bf16* __restrict__ Wt) {
  int e = blockIdx.x * 256 + threadIdx.x;   // 4 * 512*512
  int w = e >> 18;
  int r = e & 262143;
  int n = r >> 9;
  int k = r & 511;
  const float* W = (w == 0) ? Wk : (w == 1) ? Wq : (w == 2) ? Wv : Wp;
  Wt[e] = (bf16)W[k * C_ + n];             // Wt[w][n][k] = W[k][n]
}

// ---------------- kernel 1a: per-chunk angle sums (chunks of 128 in T) ------
__global__ void k_chunk_sums(const int* __restrict__ idx, const float* __restrict__ tab,
                             float* __restrict__ sums) {
  int b  = blockIdx.x >> 5;
  int ch = blockIdx.x & 31;
  int c  = threadIdx.x;                    // 0..255
  float acc = 0.f;
  int t0 = ch * 128;
  for (int t = 0; t < 128; ++t) {
    int tok = idx[b * T_ + t0 + t];
    acc += tab[(size_t)tok * C2_ + c];
  }
  sums[(b * 32 + ch) * C2_ + c] = acc;
}

// ---------------- kernel 1b: reverse cumsum within chunk + cos/sin ----------
__global__ void k_angles(const int* __restrict__ idx, const float* __restrict__ tab,
                         const float* __restrict__ sums,
                         float* __restrict__ cosb, float* __restrict__ sinb) {
  int b  = blockIdx.x >> 5;
  int ch = blockIdx.x & 31;
  int c  = threadIdx.x;
  float acc = 0.f;
  for (int ch2 = ch + 1; ch2 < 32; ++ch2) acc += sums[(b * 32 + ch2) * C2_ + c];
  int t0 = ch * 128;
  for (int t = 127; t >= 0; --t) {
    int tt  = t0 + t;
    int tok = idx[b * T_ + tt];
    acc += tab[(size_t)tok * C2_ + c];
    float s, cs;
    __sincosf(acc, &s, &cs);
    cosb[((size_t)b * T_ + tt) * C2_ + c] = cs;
    sinb[((size_t)b * T_ + tt) * C2_ + c] = s;
  }
}

// ---------------- kernel 2: fused QKV GEMM (unrotated, bf16 out) ------------
// grid (128, 8), block 256 (8 waves: 4 row-strips x 2 col-halves).
__global__ __launch_bounds__(256) void k_qkv_gemm(
    const float* __restrict__ x, const bf16* __restrict__ Wt,
    const float* __restrict__ bk, const float* __restrict__ bq, const float* __restrict__ bv,
    bf16* __restrict__ ku, bf16* __restrict__ qu, bf16* __restrict__ vu) {
  __shared__ bf16 xs[128 * 32];
  const int tid  = threadIdx.x;
  const int w    = tid >> 5, lane = tid & 31;
  const int wr   = w >> 1, wc = w & 1;
  const int m0   = blockIdx.x * 128;
  const int n0   = blockIdx.y * 64;
  f32x8 acc[3][2][2] = {};

  for (int k0 = 0; k0 < C_; k0 += 32) {
#pragma unroll
    for (int it = 0; it < 4; ++it) {       // stage 128x32 f32 -> bf16 LDS
      int idx4 = it * 256 + tid;
      int row  = idx4 >> 3;
      int c4   = (idx4 & 7) << 2;
      float4 v = *(const float4*)(x + (size_t)(m0 + row) * C_ + k0 + c4);
      bf16* d  = xs + row * 32 + c4;
      d[0] = (bf16)v.x; d[1] = (bf16)v.y; d[2] = (bf16)v.z; d[3] = (bf16)v.w;
    }
    __syncthreads();
    bf16x16 a0 = load_a_frag(xs + (wr * 32 + 0)  * 32, 32);
    bf16x16 a1 = load_a_frag(xs + (wr * 32 + 16) * 32, 32);
#pragma unroll
    for (int c = 0; c < 2; ++c) {
      const int nrow = n0 + wc * 32 + c * 16;
#pragma unroll
      for (int g = 0; g < 3; ++g) {
        bf16x16 bf = load_b_frag(Wt + (size_t)g * 262144 + (size_t)nrow * C_ + k0, C_);
        acc[g][0][c] = wmma_bf16(a0, bf, acc[g][0][c]);
        acc[g][1][c] = wmma_bf16(a1, bf, acc[g][1][c]);
      }
    }
    __syncthreads();
  }
  const int nl = lane & 15, mb = (lane >> 4) << 3;
#pragma unroll
  for (int r = 0; r < 2; ++r)
#pragma unroll
    for (int c = 0; c < 2; ++c) {
      int col = n0 + wc * 32 + c * 16 + nl;
      float b0 = bk[col], b1 = bq[col], b2 = bv[col];
#pragma unroll
      for (int g = 0; g < 8; ++g) {
        size_t row = (size_t)(m0 + wr * 32 + r * 16 + mb + g);
        ku[row * C_ + col] = (bf16)(acc[0][r][c][g] + b0);
        qu[row * C_ + col] = (bf16)(acc[1][r][c][g] + b1);
        vu[row * C_ + col] = (bf16)(acc[2][r][c][g] + b2);
      }
    }
}

// ---------------- kernel 3: rotate k,q,v; v stored transposed (c-major) -----
__global__ void k_rotate(const bf16* __restrict__ ku, const bf16* __restrict__ qu,
                         const bf16* __restrict__ vu,
                         const float* __restrict__ cosb, const float* __restrict__ sinb,
                         bf16* __restrict__ kr, bf16* __restrict__ qr, bf16* __restrict__ vT) {
  size_t e = (size_t)blockIdx.x * 256 + threadIdx.x;  // B*T*256
  size_t bt = e >> 8;
  int c2 = (int)(e & 255);
  float cs = cosb[e], sn = sinb[e];
  size_t p = bt * C_ + 2 * c2;
  float z0 = (float)ku[p], z1 = (float)ku[p + 1];
  kr[p] = (bf16)(cs * z0 + sn * z1);  kr[p + 1] = (bf16)(-sn * z0 + cs * z1);
  z0 = (float)qu[p]; z1 = (float)qu[p + 1];
  qr[p] = (bf16)(cs * z0 + sn * z1);  qr[p + 1] = (bf16)(-sn * z0 + cs * z1);
  z0 = (float)vu[p]; z1 = (float)vu[p + 1];
  int b = (int)(bt >> 12), t = (int)(bt & 4095);
  vT[((size_t)b * C_ + 2 * c2) * T_ + t]     = (bf16)(cs * z0 + sn * z1);
  vT[((size_t)b * C_ + 2 * c2 + 1) * T_ + t] = (bf16)(-sn * z0 + cs * z1);
}

// ---------------- kernel 4: streaming softplus attention -------------------
// grid (T/64, B), block 256. Workgroup owns 64 key-rows x all 512 cols of O.
// LDS (dynamic): K block 64x512 bf16 (64KB, staged async) + P block 64x64 bf16 (8KB).
__global__ __launch_bounds__(256) void k_attn(const bf16* __restrict__ kr,
                                              const bf16* __restrict__ qr,
                                              const bf16* __restrict__ vT,
                                              float* __restrict__ O) {
  extern __shared__ char smem[];
  bf16* Ks = (bf16*)smem;                 // 64*512
  bf16* Ps = (bf16*)(smem + 65536);       // 64*64
  const int b   = blockIdx.y;
  const int i0  = blockIdx.x * 64;
  const int tid = threadIdx.x;
  const int w = tid >> 5, lane = tid & 31;
  const int ti   = w >> 1;                // row strip 0..3
  const int half = w & 1;                 // col half (S: tj pair; O: 256-col half)
  const int nl = lane & 15, mb = (lane >> 4) << 3;
  const float scale = 0.044194173824159216f;  // 512^-0.5

  // stage K block (64x512 bf16) through the async engine (no VGPR round-trip)
#pragma unroll
  for (int it = 0; it < 16; ++it) {
    int c16 = it * 256 + tid;
    int row = c16 >> 5;
    int off = (c16 & 31) << 3;
    async_ld_b128(Ks + row * C_ + off,
                  kr + ((size_t)(b * T_ + i0 + row)) * C_ + off);
  }
  async_wait0();
  __syncthreads();

  f32x8 o[16] = {};                       // 16 rows x 256 cols of O per wave

  for (int j0 = 0; j0 <= i0; j0 += 64) {
    // warm L2 for the next j-block's Q while we compute this one
    if (j0 + 64 <= i0)
      __builtin_prefetch(qr + ((size_t)(b * T_ + j0 + 64)) * C_ + (size_t)tid * 128, 0, 1);

    // --- phase 1: S strip = K_i(16x512) * Q_j^T for tj in {2*half, 2*half+1}
    f32x8 s0 = {}, s1 = {};
    const bf16* qbase = qr + ((size_t)(b * T_ + j0)) * C_;
#pragma unroll
    for (int kk = 0; kk < 16; ++kk) {
      bf16x16 a  = load_a_frag(Ks + (ti * 16) * C_ + kk * 32, C_);
      bf16x16 b0 = load_b_frag(qbase + (size_t)(half * 32 + 0)  * C_ + kk * 32, C_);
      bf16x16 b1 = load_b_frag(qbase + (size_t)(half * 32 + 16) * C_ + kk * 32, C_);
      s0 = wmma_bf16(a, b0, s0);
      s1 = wmma_bf16(a, b1, s1);
    }
#pragma unroll
    for (int which = 0; which < 2; ++which) {
      int tj = half * 2 + which;
      int jc = j0 + tj * 16 + nl;
#pragma unroll
      for (int g = 0; g < 8; ++g) {
        float sv = which ? s1[g] : s0[g];
        int irow = i0 + ti * 16 + mb + g;
        float p  = (jc <= irow) ? softplusf(sv * scale) : 0.f;   // causal + softplus
        Ps[(ti * 16 + mb + g) * 64 + tj * 16 + nl] = (bf16)p;
      }
    }
    __syncthreads();
    // --- phase 2: O(16x256 strip) += P(16x64) * V_j(64x256 half)
    bf16x16 a0 = load_a_frag(Ps + (ti * 16) * 64 + 0,  64);
    bf16x16 a1 = load_a_frag(Ps + (ti * 16) * 64 + 32, 64);
    const bf16* vb = vT + ((size_t)b * C_ + half * 256) * T_ + j0;
#pragma unroll
    for (int tt = 0; tt < 16; ++tt) {
      bf16x16 b0 = load_b_frag(vb + (size_t)(tt * 16) * T_ + 0,  T_);
      bf16x16 b1 = load_b_frag(vb + (size_t)(tt * 16) * T_ + 32, T_);
      o[tt] = wmma_bf16(a0, b0, o[tt]);
      o[tt] = wmma_bf16(a1, b1, o[tt]);
    }
    __syncthreads();
  }

#pragma unroll
  for (int tt = 0; tt < 16; ++tt) {
    int col = half * 256 + tt * 16 + nl;
#pragma unroll
    for (int g = 0; g < 8; ++g) {
      int row = i0 + ti * 16 + mb + g;
      O[((size_t)(b * T_ + row)) * C_ + col] = o[tt][g];
    }
  }
}

// ---------------- kernel 5: inverse rotation -> bf16 -----------------------
__global__ void k_invrot(const float* __restrict__ O, const float* __restrict__ cosb,
                         const float* __restrict__ sinb, bf16* __restrict__ orot) {
  size_t e = (size_t)blockIdx.x * 256 + threadIdx.x;
  size_t bt = e >> 8;
  int c2 = (int)(e & 255);
  float cs = cosb[e], sn = sinb[e];
  size_t p = bt * C_ + 2 * c2;
  float z0 = O[p], z1 = O[p + 1];
  orot[p]     = (bf16)(cs * z0 - sn * z1);
  orot[p + 1] = (bf16)(sn * z0 + cs * z1);
}

// ---------------- kernel 6: final projection GEMM + bias -> f32 out --------
__global__ __launch_bounds__(256) void k_out_gemm(const bf16* __restrict__ A,
                                                  const bf16* __restrict__ Wpt,
                                                  const float* __restrict__ bp,
                                                  float* __restrict__ out) {
  __shared__ bf16 xs[128 * 32];
  const int tid = threadIdx.x;
  const int w = tid >> 5, lane = tid & 31;
  const int wr = w >> 1, wc = w & 1;
  const int m0 = blockIdx.x * 128;
  const int n0 = blockIdx.y * 64;
  f32x8 acc[2][2] = {};

  for (int k0 = 0; k0 < C_; k0 += 32) {
#pragma unroll
    for (int it = 0; it < 2; ++it) {      // stage 128x32 bf16 via async engine
      int c16 = it * 256 + tid;
      int row = c16 >> 2;
      int off = (c16 & 3) << 3;
      async_ld_b128(xs + row * 32 + off,
                    A + (size_t)(m0 + row) * C_ + k0 + off);
    }
    async_wait0();
    __syncthreads();
    bf16x16 a0 = load_a_frag(xs + (wr * 32 + 0)  * 32, 32);
    bf16x16 a1 = load_a_frag(xs + (wr * 32 + 16) * 32, 32);
#pragma unroll
    for (int c = 0; c < 2; ++c) {
      bf16x16 bf = load_b_frag(Wpt + (size_t)(n0 + wc * 32 + c * 16) * C_ + k0, C_);
      acc[0][c] = wmma_bf16(a0, bf, acc[0][c]);
      acc[1][c] = wmma_bf16(a1, bf, acc[1][c]);
    }
    __syncthreads();
  }
  const int nl = lane & 15, mb = (lane >> 4) << 3;
#pragma unroll
  for (int r = 0; r < 2; ++r)
#pragma unroll
    for (int c = 0; c < 2; ++c) {
      int col = n0 + wc * 32 + c * 16 + nl;
      float bb = bp[col];
#pragma unroll
      for (int g = 0; g < 8; ++g) {
        size_t row = (size_t)(m0 + wr * 32 + r * 16 + mb + g);
        out[row * C_ + col] = acc[r][c][g] + bb;
      }
    }
}

// ---------------------------------------------------------------------------
extern "C" void kernel_launch(void* const* d_in, const int* in_sizes, int n_in,
                              void* d_out, int out_size, void* d_ws, size_t ws_size,
                              hipStream_t stream) {
  const float* x   = (const float*)d_in[0];
  const int*   idx = (const int*)  d_in[1];
  const float* Wk  = (const float*)d_in[2];
  const float* bk  = (const float*)d_in[3];
  const float* Wq  = (const float*)d_in[4];
  const float* bq  = (const float*)d_in[5];
  const float* Wv  = (const float*)d_in[6];
  const float* bv  = (const float*)d_in[7];
  const float* Wp  = (const float*)d_in[8];
  const float* bp  = (const float*)d_in[9];
  const float* tab = (const float*)d_in[10];
  float* out = (float*)d_out;
  char*  ws  = (char*)d_ws;
  const size_t MB = 1ull << 20;

  // workspace layout (~147 MB total); O reuses the dead unrotated-k/q region.
  bf16*  Wt   = (bf16*) (ws + 0);        //  2 MB: 4x 512x512 bf16, transposed
  float* cosb = (float*)(ws + 2  * MB);  // 16 MB
  float* sinb = (float*)(ws + 18 * MB);  // 16 MB
  float* sums = (float*)(ws + 34 * MB);  // 128 KB
  bf16*  ku   = (bf16*) (ws + 35 * MB);  // 16 MB (dead after k_rotate)
  bf16*  qu   = (bf16*) (ws + 51 * MB);  // 16 MB (dead after k_rotate)
  bf16*  vu   = (bf16*) (ws + 67 * MB);  // 16 MB
  bf16*  kr   = (bf16*) (ws + 83 * MB);  // 16 MB
  bf16*  qr   = (bf16*) (ws + 99 * MB);  // 16 MB
  bf16*  vT   = (bf16*) (ws + 115 * MB); // 16 MB (channel-major)
  float* Obuf = (float*)(ws + 35 * MB);  // 32 MB, overlaps ku+qu
  bf16*  orot = (bf16*) (ws + 131 * MB); // 16 MB

  k_prep_weights<<<4096, 256, 0, stream>>>(Wk, Wq, Wv, Wp, Wt);
  k_chunk_sums  <<<128,  256, 0, stream>>>(idx, tab, sums);
  k_angles      <<<128,  256, 0, stream>>>(idx, tab, sums, cosb, sinb);
  k_qkv_gemm    <<<dim3(128, 8), 256, 0, stream>>>(x, Wt, bk, bq, bv, ku, qu, vu);
  k_rotate      <<<16384, 256, 0, stream>>>(ku, qu, vu, cosb, sinb, kr, qr, vT);
  k_attn        <<<dim3(64, 4), 256, 73728, stream>>>(kr, qr, vT, Obuf);
  k_invrot      <<<16384, 256, 0, stream>>>(Obuf, cosb, sinb, orot);
  k_out_gemm    <<<dim3(128, 8), 256, 0, stream>>>(orot, Wt + 3 * 262144, bp, out);
}
